// ChannelWiseSelfAttention_16733192585763
// MI455X (gfx1250) — matrix operand order repaired and checked by reading, not compiled, not run on hardware
//
#include <hip/hip_runtime.h>
#include <hip/hip_bf16.h>

// ---------------- problem constants ----------------
constexpr int B_  = 8;
constexpr int C_  = 192;
constexpr int NH  = 8;
constexpr int HD  = 24;
constexpr int TC  = 576;      // 3*C
constexpr int HWN = 16384;    // H*W

// ---------------- workspace layout (bytes) ----------------
constexpr size_t OFFB_WQ   = 0;                                          // 576x192 bf16
constexpr size_t OFFB_WP   = OFFB_WQ  + (size_t)TC * C_ * 2;             // 192x192 bf16
constexpr size_t OFFB_XT   = OFFB_WP  + (size_t)C_ * C_ * 2;             // (B,N,C) bf16
constexpr size_t OFFB_QK   = OFFB_XT  + (size_t)B_ * HWN * C_ * 2;       // (B,384,N) bf16
constexpr size_t OFFB_VT   = OFFB_QK  + (size_t)B_ * 384 * HWN * 2;      // (B,N,192) bf16
constexpr size_t OFFB_AVT  = OFFB_VT  + (size_t)B_ * HWN * C_ * 2;       // (B,N,192) bf16
constexpr size_t OFFB_INV  = OFFB_AVT + (size_t)B_ * HWN * C_ * 2;       // (B,384) f32
constexpr size_t OFFB_ATTN = OFFB_INV + (size_t)B_ * 384 * 4;            // (B,NH,32,32) bf16

// ---------------- types / helpers ----------------
typedef __attribute__((ext_vector_type(16))) __bf16 v16bf;
typedef __attribute__((ext_vector_type(8)))  __bf16 v8bf;
typedef __attribute__((ext_vector_type(8)))  float  v8f;
typedef __attribute__((ext_vector_type(8)))  unsigned short u16x8;

static __device__ __forceinline__ __bf16 f2bf(float f) {
  unsigned u = __builtin_bit_cast(unsigned, f);
  u += 0x7FFFu + ((u >> 16) & 1u);            // round-to-nearest-even
  unsigned short s = (unsigned short)(u >> 16);
  return __builtin_bit_cast(__bf16, s);
}
static __device__ __forceinline__ float bf2f(__bf16 h) {
  unsigned short s = __builtin_bit_cast(unsigned short, h);
  unsigned u = ((unsigned)s) << 16;
  return __builtin_bit_cast(float, u);
}
static __device__ __forceinline__ v8bf zero8() {
  u16x8 z = (u16x8)0;
  return __builtin_bit_cast(v8bf, z);
}
static __device__ __forceinline__ v16bf cat8(v8bf lo, v8bf hi) {
  return __builtin_shufflevector(lo, hi, 0,1,2,3,4,5,6,7,8,9,10,11,12,13,14,15);
}
static __device__ __forceinline__ v8f wmma_bf16(v16bf a, v16bf b, v8f c) {
  return __builtin_amdgcn_wmma_f32_16x16x32_bf16(false, a, false, b, (short)0, c, false, false);
}
// Async copy of 16 bytes from global to LDS (per lane). lds_off = low 32 bits of
// the flat shared-pointer (ISA: flat->LDS mapping truncates to addr[31:0]).
static __device__ __forceinline__ void async_g2l_b128(unsigned lds_off, const void* gptr) {
  asm volatile("global_load_async_to_lds_b128 %0, %1, off"
               :: "v"(lds_off), "v"((unsigned long long)(uintptr_t)gptr)
               : "memory");
}

// ================= kernel 0: weight transpose + bf16 convert =================
__global__ void prep_weights_kernel(const float* __restrict__ Wqkv,
                                    const float* __restrict__ Wproj,
                                    __bf16* __restrict__ wq_t,
                                    __bf16* __restrict__ wp_t) {
  int id = blockIdx.x * 256 + threadIdx.x;
  if (id < TC * C_) {
    int oc = id / C_, c = id % C_;
    wq_t[id] = f2bf(Wqkv[(size_t)c * TC + oc]);       // Wt[oc][c] = W[c][oc]
  }
  int id2 = id - TC * C_;
  if (id2 >= 0 && id2 < C_ * C_) {
    int oc = id2 / C_, c = id2 % C_;
    wp_t[id2] = f2bf(Wproj[(size_t)c * C_ + oc]);
  }
}

// ================= kernel 1: x (B,C,N) f32 -> xt (B,N,C) bf16 =================
__global__ void xt_transpose_kernel(const float* __restrict__ x,
                                    __bf16* __restrict__ xt) {
  __shared__ float tile[32][68];
  int b = blockIdx.z, c0 = blockIdx.y * 32, n0 = blockIdx.x * 64;
  int t = threadIdx.x;
  {
    int r = t >> 3, cb = (t & 7) * 8;                 // 32 c-rows x 64 n-cols
    const float* src = x + ((size_t)(b * C_ + c0 + r)) * HWN + n0 + cb;
    float4 f0 = *(const float4*)(src);
    float4 f1 = *(const float4*)(src + 4);
    tile[r][cb + 0] = f0.x; tile[r][cb + 1] = f0.y;
    tile[r][cb + 2] = f0.z; tile[r][cb + 3] = f0.w;
    tile[r][cb + 4] = f1.x; tile[r][cb + 5] = f1.y;
    tile[r][cb + 6] = f1.z; tile[r][cb + 7] = f1.w;
  }
  __syncthreads();
  {
    int n = t >> 2, g = (t & 3) * 8;                  // 64 n-rows x 32 c-cols
    v8bf v;
    #pragma unroll
    for (int k = 0; k < 8; k++) v[k] = f2bf(tile[g + k][n]);
    *(v8bf*)(xt + ((size_t)(b * HWN) + n0 + n) * C_ + c0 + g) = v;
  }
}

// ================= kernel 2: QKV GEMM (async-LDS staged B operand) ==========
// D[b,oc,n] = bias[oc] + sum_c Wt[oc,c] * xt[b,n,c]
// oc<384 -> qk channel-major; oc>=384 -> vt n-major
__global__ void qkv_gemm_kernel(const __bf16* __restrict__ wq_t,
                                const float* __restrict__ bias,
                                const __bf16* __restrict__ xt,
                                __bf16* __restrict__ qk,
                                __bf16* __restrict__ vt) {
  __shared__ __bf16 bs[2][128 * 32];                  // double-buffered 8KB B tiles
  int b  = blockIdx.z;
  int m0 = blockIdx.y * 64;
  int n0 = blockIdx.x * 128;
  int tid = threadIdx.x, lane = tid & 31, wid = tid >> 5;
  int hf = lane >> 4, l15 = lane & 15;
  int wm = wid & 3, wn = wid >> 2;
  int row_m = m0 + wm * 16 + l15;
  const __bf16* arow = wq_t + (size_t)row_m * C_;
  __builtin_prefetch(arow, 0, 0);                      // global_prefetch weight panel

  // per-thread async-copy mapping: 16 bf16 (16B? no: 16 elems = 32B -> one b128 is 16B)
  // thread t copies 8 elems? -> we copy 16B (8 bf16) ... use: n = t & 127, kq = t>>7 (16 elems? )
  // Layout: each thread copies 16 bytes = 8 bf16.  Tile = 128n x 32k -> 8KB = 512 x 16B,
  // so each of 256 threads issues two b128 copies (k-quarters).
  int sn = tid & 127;                                  // n row 0..127
  int sq = tid >> 7;                                   // 0/1 -> k quarter pair (k 16*sq..16*sq+15)
  const __bf16* gsrc = xt + ((size_t)b * HWN + n0 + sn) * C_;
  unsigned ldsrow0 = (unsigned)(uintptr_t)&bs[0][sn * 32];
  unsigned ldsrow1 = (unsigned)(uintptr_t)&bs[1][sn * 32];

  v8f acc[4] = {};
  // prologue: stage K-tile 0 into buffer 0 (each thread: 2 x 16B)
  async_g2l_b128(ldsrow0 + (unsigned)(sq * 16) * 2, gsrc + 0 + sq * 16);
  async_g2l_b128(ldsrow0 + (unsigned)(sq * 16 + 8) * 2, gsrc + 0 + sq * 16 + 8);

  #pragma unroll
  for (int s = 0; s < 6; s++) {
    int buf = s & 1;
    if (s < 5) {
      unsigned dst = (buf ? ldsrow0 : ldsrow1) + (unsigned)(sq * 16) * 2;
      const __bf16* g = gsrc + (s + 1) * 32 + sq * 16;
      async_g2l_b128(dst, g);
      async_g2l_b128(dst + 16, g + 8);
      asm volatile("s_wait_asynccnt 0x2" ::: "memory");   // my previous tile done
    } else {
      asm volatile("s_wait_asynccnt 0x0" ::: "memory");
    }
    __syncthreads();                                      // everyone's tile s visible

    int kk = s * 32;
    v16bf a = cat8(*(const v8bf*)(arow + kk + hf * 16),
                   *(const v8bf*)(arow + kk + hf * 16 + 8));
    #pragma unroll
    for (int nn = 0; nn < 4; nn++) {
      int nl = wn * 64 + nn * 16 + l15;
      v16bf bf = *(const v16bf*)(&bs[buf][nl * 32 + hf * 16]);  // ds_load_b128 x2
      acc[nn] = wmma_bf16(a, bf, acc[nn]);
    }
    __syncthreads();                                      // protect buf from overwrite
  }

  #pragma unroll
  for (int nn = 0; nn < 4; nn++) {
    int n = n0 + wn * 64 + nn * 16 + l15;
    #pragma unroll
    for (int r = 0; r < 8; r++) {
      int m = m0 + wm * 16 + hf * 8 + r;              // D: lanes16-31 hold M=8+r
      float v = acc[nn][r] + bias[m];
      if (m < 384) qk[((size_t)b * 384 + m) * HWN + n] = f2bf(v);
      else         vt[((size_t)b * HWN + n) * C_ + (m - 384)] = f2bf(v);
    }
  }
}

// ================= kernel 3: 1/||q_ch||, 1/||k_ch|| over N =================
__global__ void norm_kernel(const __bf16* __restrict__ qk,
                            float* __restrict__ inv) {
  int id = blockIdx.x;                                // b*384 + ch
  const v8bf* p = (const v8bf*)(qk + (size_t)id * HWN);
  float s = 0.f;
  for (int i = threadIdx.x; i < HWN / 8; i += 256) {
    v8bf v = p[i];
    #pragma unroll
    for (int k = 0; k < 8; k++) { float f = bf2f(v[k]); s += f * f; }
  }
  #pragma unroll
  for (int o = 16; o > 0; o >>= 1) s += __shfl_xor(s, o, 32);
  __shared__ float partial[8];
  if ((threadIdx.x & 31) == 0) partial[threadIdx.x >> 5] = s;
  __syncthreads();
  if (threadIdx.x == 0) {
    float t = 0.f;
    #pragma unroll
    for (int i = 0; i < 8; i++) t += partial[i];
    inv[id] = 1.0f / fmaxf(sqrtf(t), 1e-12f);
  }
}

// ================= kernel 4: logits (WMMA, K split over waves) + softmax ====
__global__ void attn_softmax_kernel(const __bf16* __restrict__ qk,
                                    const float* __restrict__ inv,
                                    const float* __restrict__ temp,
                                    __bf16* __restrict__ attn) {
  int h = blockIdx.x, b = blockIdx.y;
  int tid = threadIdx.x, lane = tid & 31, w = tid >> 5;
  int hf = lane >> 4, l15 = lane & 15;
  size_t qbase = ((size_t)b * 384 + h * HD) * HWN;
  size_t kbase = ((size_t)b * 384 + 192 + h * HD) * HWN;
  int kb0 = w * (HWN / 8);
  v8f acc[2][2] = {};
  for (int s = 0; s < (HWN / 8) / 32; s++) {
    int kk = kb0 + s * 32;
    v16bf afr[2], bfr[2];
    #pragma unroll
    for (int mi = 0; mi < 2; mi++) {
      int row = mi * 16 + l15;
      if (row < HD) {
        const __bf16* p = qk + qbase + (size_t)row * HWN + kk;
        afr[mi] = cat8(*(const v8bf*)(p + hf * 16), *(const v8bf*)(p + hf * 16 + 8));
      } else afr[mi] = cat8(zero8(), zero8());
    }
    #pragma unroll
    for (int ni = 0; ni < 2; ni++) {
      int col = ni * 16 + l15;
      if (col < HD) bfr[ni] = *(const v16bf*)(qk + kbase + (size_t)col * HWN + kk + hf * 16);
      else          bfr[ni] = cat8(zero8(), zero8());
    }
    #pragma unroll
    for (int mi = 0; mi < 2; mi++)
      #pragma unroll
      for (int ni = 0; ni < 2; ni++)
        acc[mi][ni] = wmma_bf16(afr[mi], bfr[ni], acc[mi][ni]);
  }
  __shared__ float red[8 * 1024];
  __shared__ float logits[32 * 32];
  #pragma unroll
  for (int mi = 0; mi < 2; mi++)
    #pragma unroll
    for (int ni = 0; ni < 2; ni++)
      #pragma unroll
      for (int r = 0; r < 8; r++)
        red[w * 1024 + (mi * 2 + ni) * 256 + r * 32 + lane] = acc[mi][ni][r];
  __syncthreads();
  for (int t = tid; t < 1024; t += 256) {
    float s = 0.f;
    #pragma unroll
    for (int ww = 0; ww < 8; ww++) s += red[ww * 1024 + t];
    int tile = t >> 8, r = (t >> 5) & 7, ln = t & 31;
    int i = (tile >> 1) * 16 + (ln >> 4) * 8 + r;
    int j = (tile & 1) * 16 + (ln & 15);
    if (i < HD && j < HD)
      logits[i * 32 + j] = s * temp[h] * inv[b * 384 + h * HD + i]
                             * inv[b * 384 + 192 + h * HD + j];
  }
  __syncthreads();
  __bf16* ap = attn + (size_t)(b * NH + h) * 1024;    // 32x32, [i][j], zero padded
  #pragma unroll
  for (int rr = 0; rr < 4; rr++) {
    int i = w + rr * 8;
    float p = 0.f;
    if (i < HD) {
      float val = (lane < HD) ? logits[i * 32 + lane] : -3.0e38f;
      float m = val;
      #pragma unroll
      for (int o = 16; o > 0; o >>= 1) m = fmaxf(m, __shfl_xor(m, o, 32));
      float e = (lane < HD) ? __expf(val - m) : 0.f;
      float ssum = e;
      #pragma unroll
      for (int o = 16; o > 0; o >>= 1) ssum += __shfl_xor(ssum, o, 32);
      p = e / ssum;
    }
    ap[i * 32 + lane] = f2bf(p);
  }
}

// ================= kernel 5: out^T[n,i] = sum_j vt[n,j]*attn[i,j] ==========
__global__ void av_kernel(const __bf16* __restrict__ vt,
                          const __bf16* __restrict__ attn,
                          __bf16* __restrict__ avt) {
  int b = blockIdx.z, h = blockIdx.y;
  int n0 = blockIdx.x * 128;
  int tid = threadIdx.x, lane = tid & 31, w = tid >> 5;
  int hf = lane >> 4, l15 = lane & 15;
  int n = n0 + w * 16 + l15;
  const __bf16* vrow = vt + ((size_t)b * HWN + n) * C_ + h * HD;
  // A fragment over K=j (padded to 32): hi half of lanes16-31 is j=24..31 -> zero
  v8bf alo = *(const v8bf*)(vrow + hf * 8);
  v8bf ahi = hf ? zero8() : *(const v8bf*)(vrow + 16);
  v16bf a = cat8(alo, ahi);
  const __bf16* ap = attn + (size_t)(b * NH + h) * 1024;
  v8f acc[2] = {};
  #pragma unroll
  for (int ni = 0; ni < 2; ni++) {
    int i = ni * 16 + l15;
    v16bf bf = cat8(*(const v8bf*)(ap + i * 32 + hf * 8), *(const v8bf*)(ap + i * 32 + 16 + hf * 8));   // B[j,i] = attn[i][j], table-order j
    acc[ni] = wmma_bf16(a, bf, acc[ni]);
  }
  #pragma unroll
  for (int ni = 0; ni < 2; ni++) {
    int col = ni * 16 + l15;
    if (col < HD) {
      #pragma unroll
      for (int r = 0; r < 8; r++) {
        int m = n0 + w * 16 + hf * 8 + r;
        avt[((size_t)b * HWN + m) * C_ + h * HD + col] = f2bf(acc[ni][r]);
      }
    }
  }
}

// ================= kernel 6: projection GEMM (async-LDS staged B) ==========
__global__ void proj_gemm_kernel(const __bf16* __restrict__ wp_t,
                                 const float* __restrict__ bias,
                                 const __bf16* __restrict__ avt,
                                 float* __restrict__ out) {
  __shared__ __bf16 bs[2][128 * 32];
  int b  = blockIdx.z;
  int m0 = blockIdx.y * 64;
  int n0 = blockIdx.x * 128;
  int tid = threadIdx.x, lane = tid & 31, wid = tid >> 5;
  int hf = lane >> 4, l15 = lane & 15;
  int wm = wid & 3, wn = wid >> 2;
  int row_m = m0 + wm * 16 + l15;
  const __bf16* arow = wp_t + (size_t)row_m * C_;
  __builtin_prefetch(arow, 0, 0);

  int sn = tid & 127;
  int sq = tid >> 7;
  const __bf16* gsrc = avt + ((size_t)b * HWN + n0 + sn) * C_;
  unsigned ldsrow0 = (unsigned)(uintptr_t)&bs[0][sn * 32];
  unsigned ldsrow1 = (unsigned)(uintptr_t)&bs[1][sn * 32];

  v8f acc[4] = {};
  async_g2l_b128(ldsrow0 + (unsigned)(sq * 16) * 2, gsrc + 0 + sq * 16);
  async_g2l_b128(ldsrow0 + (unsigned)(sq * 16 + 8) * 2, gsrc + 0 + sq * 16 + 8);

  #pragma unroll
  for (int s = 0; s < 6; s++) {
    int buf = s & 1;
    if (s < 5) {
      unsigned dst = (buf ? ldsrow0 : ldsrow1) + (unsigned)(sq * 16) * 2;
      const __bf16* g = gsrc + (s + 1) * 32 + sq * 16;
      async_g2l_b128(dst, g);
      async_g2l_b128(dst + 16, g + 8);
      asm volatile("s_wait_asynccnt 0x2" ::: "memory");
    } else {
      asm volatile("s_wait_asynccnt 0x0" ::: "memory");
    }
    __syncthreads();

    int kk = s * 32;
    v16bf a = cat8(*(const v8bf*)(arow + kk + hf * 16),
                   *(const v8bf*)(arow + kk + hf * 16 + 8));
    #pragma unroll
    for (int nn = 0; nn < 4; nn++) {
      int nl = wn * 64 + nn * 16 + l15;
      v16bf bf = *(const v16bf*)(&bs[buf][nl * 32 + hf * 16]);
      acc[nn] = wmma_bf16(a, bf, acc[nn]);
    }
    __syncthreads();
  }

  #pragma unroll
  for (int nn = 0; nn < 4; nn++) {
    int n = n0 + wn * 64 + nn * 16 + l15;
    #pragma unroll
    for (int r = 0; r < 8; r++) {
      int m = m0 + wm * 16 + hf * 8 + r;
      out[((size_t)b * C_ + m) * HWN + n] = acc[nn][r] + bias[m];
    }
  }
}

// ================= host launcher =================
extern "C" void kernel_launch(void* const* d_in, const int* in_sizes, int n_in,
                              void* d_out, int out_size, void* d_ws, size_t ws_size,
                              hipStream_t stream) {
  const float* x      = (const float*)d_in[0];
  const float* W_qkv  = (const float*)d_in[1];
  const float* b_qkv  = (const float*)d_in[2];
  const float* temp   = (const float*)d_in[3];
  const float* W_proj = (const float*)d_in[4];
  const float* b_proj = (const float*)d_in[5];
  float* out = (float*)d_out;

  char* ws = (char*)d_ws;
  __bf16* wq_t = (__bf16*)(ws + OFFB_WQ);
  __bf16* wp_t = (__bf16*)(ws + OFFB_WP);
  __bf16* xt   = (__bf16*)(ws + OFFB_XT);
  __bf16* qk   = (__bf16*)(ws + OFFB_QK);
  __bf16* vt   = (__bf16*)(ws + OFFB_VT);
  __bf16* avt  = (__bf16*)(ws + OFFB_AVT);
  float*  inv  = (float*)(ws + OFFB_INV);
  __bf16* attn = (__bf16*)(ws + OFFB_ATTN);

  prep_weights_kernel<<<(TC * C_ + C_ * C_ + 255) / 256, 256, 0, stream>>>(W_qkv, W_proj, wq_t, wp_t);
  xt_transpose_kernel<<<dim3(HWN / 64, C_ / 32, B_), 256, 0, stream>>>(x, xt);
  qkv_gemm_kernel<<<dim3(HWN / 128, TC / 64, B_), 256, 0, stream>>>(wq_t, b_qkv, xt, qk, vt);
  norm_kernel<<<B_ * 384, 256, 0, stream>>>(qk, inv);
  attn_softmax_kernel<<<dim3(NH, B_), 256, 0, stream>>>(qk, inv, temp, attn);
  av_kernel<<<dim3(HWN / 128, NH, B_), 256, 0, stream>>>(vt, attn, avt);
  proj_gemm_kernel<<<dim3(HWN / 128, C_ / 64, B_), 256, 0, stream>>>(wp_t, b_proj, avt, out);
}
